// Attention_28321014350364
// MI455X (gfx1250) — matrix operand order repaired and checked by reading, not compiled, not run on hardware
//
#include <hip/hip_runtime.h>

// ---------------------------------------------------------------------------
// Gumbel-softmax attention for MI455X (gfx1250, wave32, WMMA).
//   out    = softmax(Q K^T / sqrt(64) + gumbel) @ V
//   d_out  = [ out (16384x1024 f32) | p_attn (16384x4096 f32) ]
// ---------------------------------------------------------------------------

typedef __attribute__((ext_vector_type(16))) _Float16 v16h;
typedef __attribute__((ext_vector_type(8)))  float    v8f;

#define G_GENES 16384
#define N_LAT   4096
#define D_K     64
#define N_CELL  1024
#define SROW    4100   // padded LDS row stride (floats) to avoid bank conflicts

#if defined(__gfx1250__)
#define HAS_ASYNC_LDS __has_builtin(__builtin_amdgcn_global_load_async_to_lds_b32)
#else
#define HAS_ASYNC_LDS 0
#endif

typedef __attribute__((address_space(1))) int GInt;   // global  (as1)
typedef __attribute__((address_space(3))) int LInt;   // LDS     (as3)

__device__ __forceinline__ void wait_async_zero() {
#if defined(__gfx1250__)
#if __has_builtin(__builtin_amdgcn_s_wait_asynccnt)
    __builtin_amdgcn_s_wait_asynccnt(0);
#else
    asm volatile("s_wait_asynccnt 0" ::: "memory");
#endif
#endif
}

// -------------------------- deterministic Gumbel ---------------------------
__device__ __forceinline__ float gumbel_of(unsigned idx) {
    unsigned h = idx;
    h ^= h >> 16; h *= 0x85ebca6bu;
    h ^= h >> 13; h *= 0xc2b2ae35u;
    h ^= h >> 16;
    float u = (float)(h >> 8) * (1.0f / 16777216.0f) + 1e-12f;
    return -__logf(-__logf(u));
}

// ---------------------------------------------------------------------------
// Kernel T: V[4096][1024] f32  ->  Vt[1024][4096] f16 (for WMMA B fragments)
// Load phase uses CDNA5 async global->LDS copies when available.
// ---------------------------------------------------------------------------
__global__ __launch_bounds__(256) void transpose_v_kernel(
    const float* __restrict__ v, _Float16* __restrict__ vt) {
    __shared__ float tile[32][33];
    const int l0 = blockIdx.y * 32;           // latent (row of V)
    const int c0 = blockIdx.x * 32;           // cell   (col of V)
    const int tx = threadIdx.x & 31;
    const int ty = threadIdx.x >> 5;          // 0..7
#if HAS_ASYNC_LDS
#pragma unroll
    for (int s = 0; s < 4; ++s) {
        const float* g = v + (size_t)(l0 + ty + 8 * s) * N_CELL + (c0 + tx);
        __builtin_amdgcn_global_load_async_to_lds_b32(
            (GInt*)g, (LInt*)&tile[ty + 8 * s][tx], 0, 0);
    }
    wait_async_zero();
#else
#pragma unroll
    for (int s = 0; s < 4; ++s)
        tile[ty + 8 * s][tx] = v[(size_t)(l0 + ty + 8 * s) * N_CELL + (c0 + tx)];
#endif
    __syncthreads();
#pragma unroll
    for (int s = 0; s < 4; ++s)
        vt[(size_t)(c0 + ty + 8 * s) * N_LAT + (l0 + tx)] =
            (_Float16)tile[tx][ty + 8 * s];
}

// ---------------------------------------------------------------------------
// Kernel 1: scores = QK^T/8 + gumbel, row softmax over 4096, write p_attn.
// One block = 16 gene rows, 8 waves; scores tile lives in 262KB dynamic LDS.
// ---------------------------------------------------------------------------
__global__ __launch_bounds__(256) void qk_softmax_kernel(
    const float* __restrict__ q, const float* __restrict__ kmat,
    float* __restrict__ p) {
    extern __shared__ float smem[];
    float* sc     = smem;                       // [16][SROW]
    float* red    = smem + 16 * SROW;           // [16][16] partials
    float* rowinv = red + 256;                  // [16]

    const int tid   = threadIdx.x;
    const int lane  = tid & 31;
    const int wave  = tid >> 5;
    const int gbase = blockIdx.x * 16;

    // ---- A fragment: Q rows gbase..gbase+15, f32 -> f16 (16x32 layout x2) ----
    const int   mrow  = lane & 15;
    const int   khalf = (lane & 16) ? 8 : 0;
    const float* qrow = q + (size_t)(gbase + mrow) * D_K;
    v8f qa0 = *(const v8f*)(qrow + khalf);
    v8f qa1 = *(const v8f*)(qrow + 16 + khalf);
    v8f qa2 = *(const v8f*)(qrow + 32 + khalf);
    v8f qa3 = *(const v8f*)(qrow + 48 + khalf);
    v16h a0, a1;
#pragma unroll
    for (int h = 0; h < 8; ++h) {
        a0[h]     = (_Float16)qa0[h];
        a0[h + 8] = (_Float16)qa1[h];
        a1[h]     = (_Float16)qa2[h];
        a1[h + 8] = (_Float16)qa3[h];
    }

    const int nb  = lane & 15;
    const int kb  = (lane & 16) ? 8 : 0;
    const int mhi = (lane & 16) ? 8 : 0;

    // ---- 32 tiles of 16 latent columns per wave (8 waves * 512 = 4096) ----
    for (int t = 0; t < 32; ++t) {
        const int colbase = wave * 512 + t * 16;
        const float* krow = kmat + (size_t)(colbase + nb) * D_K + kb;
        v8f k0 = *(const v8f*)(krow);
        v8f k1 = *(const v8f*)(krow + 16);
        v8f k2 = *(const v8f*)(krow + 32);
        v8f k3 = *(const v8f*)(krow + 48);
        v16h b0, b1;
#pragma unroll
        for (int h = 0; h < 8; ++h) {
            b0[h]     = (_Float16)k0[h];
            b0[h + 8] = (_Float16)k1[h];
            b1[h]     = (_Float16)k2[h];
            b1[h + 8] = (_Float16)k3[h];
        }
        v8f c = {};
        c = __builtin_amdgcn_wmma_f32_16x16x32_f16(false, a0, false, b0,
                                                   (short)0, c, false, false);
        c = __builtin_amdgcn_wmma_f32_16x16x32_f16(false, a1, false, b1,
                                                   (short)0, c, false, false);
        const int ncol = colbase + (lane & 15);
#pragma unroll
        for (int r = 0; r < 8; ++r) {
            const int m = r + mhi;
            float s = c[r] * 0.125f;   // 1/sqrt(64)
            s += gumbel_of((unsigned)(gbase + m) * (unsigned)N_LAT +
                           (unsigned)ncol);
            sc[m * SROW + ncol] = s;
        }
    }
    __syncthreads();

    // ---- softmax: 16 threads per row ----
    const int row = tid >> 4;
    const int sub = tid & 15;
    float mx = -3.4e38f;
    for (int i = sub; i < N_LAT; i += 16) mx = fmaxf(mx, sc[row * SROW + i]);
    red[row * 16 + sub] = mx;
    __syncthreads();
    mx = red[row * 16];
#pragma unroll
    for (int i = 1; i < 16; ++i) mx = fmaxf(mx, red[row * 16 + i]);
    __syncthreads();

    float sum = 0.f;
    for (int i = sub; i < N_LAT; i += 16) {
        float e = __expf(sc[row * SROW + i] - mx);
        sc[row * SROW + i] = e;
        sum += e;
    }
    red[row * 16 + sub] = sum;
    __syncthreads();
    if (sub == 0) {
        float s = 0.f;
#pragma unroll
        for (int i = 0; i < 16; ++i) s += red[row * 16 + i];
        rowinv[row] = 1.0f / s;
    }
    __syncthreads();

    // ---- fully coalesced normalized store ----
    float* prow = p + (size_t)gbase * N_LAT;
    for (int i = tid; i < 16 * N_LAT; i += 256) {
        const int r = i >> 12;          // N_LAT == 4096 == 2^12
        const int c = i & (N_LAT - 1);
        prow[i] = sc[r * SROW + c] * rowinv[r];
    }
}

// ---------------------------------------------------------------------------
// Kernel 2: out[16384x1024] = P[16384x4096] @ V, B from pre-transposed Vt f16.
// Block = 128M x 128N; 8 waves (4M x 2N); each wave: 32M x 64N register tile
// = 2x4 accumulators -> 8 independent WMMAs per K-step (no dep-NOP stalls).
// ---------------------------------------------------------------------------
__global__ __launch_bounds__(256) void pv_gemm_kernel(
    const float* __restrict__ p, const _Float16* __restrict__ vt,
    float* __restrict__ out) {
    const int tid  = threadIdx.x;
    const int lane = tid & 31;
    const int wave = tid >> 5;
    const int m0 = blockIdx.y * 128 + (wave & 3) * 32;   // 2 x 16 rows
    const int n0 = blockIdx.x * 128 + (wave >> 2) * 64;  // 4 x 16 cols

    const int   mrow = lane & 15;
    const int   ka   = (lane & 16) ? 16 : 0;   // A half base (K)
    const int   kb   = (lane & 16) ? 16 : 0;   // B half base (K)
    const float*    arow0 = p + (size_t)(m0 + mrow) * N_LAT;
    const float*    arow1 = arow0 + (size_t)16 * N_LAT;
    const _Float16* brow0 = vt + (size_t)(n0 + (lane & 15)) * N_LAT + kb;

    v8f acc[2][4] = {{v8f{}, v8f{}, v8f{}, v8f{}},
                     {v8f{}, v8f{}, v8f{}, v8f{}}};

    for (int kt = 0; kt < N_LAT; kt += 32) {
        __builtin_prefetch((const void*)(arow0 + kt + 256), 0, 1);
        __builtin_prefetch((const void*)(arow1 + kt + 256), 0, 1);
        v16h a[2];
#pragma unroll
        for (int mt = 0; mt < 2; ++mt) {
            const float* ar = (mt == 0) ? arow0 : arow1;
            v8f af0 = *(const v8f*)(ar + kt + ka);
            v8f af1 = *(const v8f*)(ar + kt + 8 + ka);
#pragma unroll
            for (int h = 0; h < 8; ++h) {
                a[mt][h]     = (_Float16)af0[h];
                a[mt][h + 8] = (_Float16)af1[h];
            }
        }
#pragma unroll
        for (int nt = 0; nt < 4; ++nt) {
            v16h b = *(const v16h*)(brow0 + (size_t)nt * 16 * N_LAT + kt);
#pragma unroll
            for (int mt = 0; mt < 2; ++mt) {
                acc[mt][nt] = __builtin_amdgcn_wmma_f32_16x16x32_f16(
                    false, a[mt], false, b, (short)0, acc[mt][nt], false, false);
            }
        }
    }

    const int mhi = (lane & 16) ? 8 : 0;
    const int nc  = lane & 15;
#pragma unroll
    for (int mt = 0; mt < 2; ++mt) {
#pragma unroll
        for (int nt = 0; nt < 4; ++nt) {
#pragma unroll
            for (int r = 0; r < 8; ++r)
                out[(size_t)(m0 + mt * 16 + r + mhi) * N_CELL +
                    n0 + nt * 16 + nc] = acc[mt][nt][r];
        }
    }
}

// ---------------------------------------------------------------------------
extern "C" void kernel_launch(void* const* d_in, const int* in_sizes, int n_in,
                              void* d_out, int out_size, void* d_ws,
                              size_t ws_size, hipStream_t stream) {
    (void)in_sizes; (void)n_in; (void)out_size; (void)ws_size;
    const float* q = (const float*)d_in[0];   // [16384 x 64]
    const float* k = (const float*)d_in[1];   // [4096 x 64]
    const float* v = (const float*)d_in[2];   // [4096 x 1024]

    float* out = (float*)d_out;                          // [16384 x 1024]
    float* p   = out + (size_t)G_GENES * N_CELL;         // [16384 x 4096]
    _Float16* vt = (_Float16*)d_ws;                      // [1024 x 4096] f16

    // 1) V -> Vt (f16) for contiguous WMMA B fragments (async global->LDS)
    transpose_v_kernel<<<dim3(N_CELL / 32, N_LAT / 32), 256, 0, stream>>>(v, vt);

    // 2) scores + gumbel + softmax -> p_attn (262KB dynamic LDS per WG)
    const size_t smem1 = (size_t)(16 * SROW + 256 + 16) * sizeof(float);
    qk_softmax_kernel<<<dim3(G_GENES / 16), 256, smem1, stream>>>(q, k, p);

    // 3) out = p_attn @ V  (128x128 block tile, 8 WMMAs per wave per K-step)
    pv_gemm_kernel<<<dim3(N_CELL / 128, G_GENES / 128), 256, 0, stream>>>(p, vt, out);
}
